// MambaBlock_51745765982530
// MI455X (gfx1250) — matrix operand order repaired and checked by reading, compiled, not run on hardware
//
#include <hip/hip_runtime.h>
#include <hip/hip_bf16.h>
#include <math.h>

// ---------------- problem constants (from reference) ----------------
constexpr int Bn      = 2;
constexpr int Ln      = 1024;
constexpr int DMODEL  = 1024;
constexpr int DSTATE  = 16;
constexpr int DCONV   = 4;
constexpr int DINNER  = 2048;            // DMODEL * 2
constexpr int DTRANK  = 64;              // (1024+15)/16
constexpr int NPROJ   = DTRANK + 2 * DSTATE + DINNER;  // 2144
constexpr int ROWS    = Bn * Ln;         // 2048 token rows

typedef __attribute__((ext_vector_type(16))) __bf16 v16bf;
typedef __attribute__((ext_vector_type(8)))  __bf16 v8bf;
typedef __attribute__((ext_vector_type(8)))  float  v8f;

// =====================================================================
// bf16 WMMA GEMM:  C[m,n] = sum_k A[m,k] * W[n,k]
// A: row-major bf16 [M x K], lda    (pre-converted)
// W: row-major bf16 [N x K], ldb    (pre-converted; col n of B = row n of W)
// C: row-major f32  [M x N], ldc
// One wave32 computes a 32x32 output tile: 2 A frags x 2 B frags -> 4 WMMA.
// All fragment loads are contiguous b128s; A and B both use k = 16*half + s order.
// =====================================================================
__global__ __launch_bounds__(256)
void wmma_gemm_bf16(const __bf16* __restrict__ A, int lda,
                    const __bf16* __restrict__ W, int ldb,
                    float* __restrict__ C, int ldc,
                    int M, int N, int K)
{
    const int lane = threadIdx.x & 31;
    const int waveInBlock = threadIdx.x >> 5;
    const int gw = blockIdx.x * (blockDim.x >> 5) + waveInBlock;

    const int tilesN = N >> 5;                 // 32-wide N tiles
    const int totalTiles = (M >> 5) * tilesN;  // 32-tall M tiles
    if (gw >= totalTiles) return;              // wave-uniform: EXEC stays all-1s

    const int tileM = (gw / tilesN) << 5;
    const int tileN = (gw % tilesN) << 5;

    const int half = lane >> 4;                // 0: lanes 0-15, 1: lanes 16-31
    const int l16  = lane & 15;

    v8f c00 = {}, c01 = {}, c10 = {}, c11 = {};

    // A fragment base: lane holds M=l16 (+16 for second frag), 16 contiguous
    // K values at half*16 (two 8-element runs at koff and koff+8).
    const __bf16* pA0 = A + (size_t)(tileM + l16)      * lda + half * 16;
    const __bf16* pA1 = A + (size_t)(tileM + 16 + l16) * lda + half * 16;
    // B fragment base: lane holds N=l16 (+16), 16 contiguous K values at half*16.
    const __bf16* pB0 = W + (size_t)(tileN + l16)      * ldb + half * 16;
    const __bf16* pB1 = W + (size_t)(tileN + 16 + l16) * ldb + half * 16;

    for (int k = 0; k < K; k += 32) {
        v8bf a0lo = *reinterpret_cast<const v8bf*>(pA0 + k);
        v8bf a0hi = *reinterpret_cast<const v8bf*>(pA0 + k + 8);
        v8bf a1lo = *reinterpret_cast<const v8bf*>(pA1 + k);
        v8bf a1hi = *reinterpret_cast<const v8bf*>(pA1 + k + 8);
        v16bf a0 = __builtin_shufflevector(a0lo, a0hi,
                       0,1,2,3,4,5,6,7,8,9,10,11,12,13,14,15);
        v16bf a1 = __builtin_shufflevector(a1lo, a1hi,
                       0,1,2,3,4,5,6,7,8,9,10,11,12,13,14,15);
        v16bf b0 = *reinterpret_cast<const v16bf*>(pB0 + k);
        v16bf b1 = *reinterpret_cast<const v16bf*>(pB1 + k);

        c00 = __builtin_amdgcn_wmma_f32_16x16x32_bf16(false, a0, false, b0,
                                                      (short)0, c00, false, false);
        c01 = __builtin_amdgcn_wmma_f32_16x16x32_bf16(false, a0, false, b1,
                                                      (short)0, c01, false, false);
        c10 = __builtin_amdgcn_wmma_f32_16x16x32_bf16(false, a1, false, b0,
                                                      (short)0, c10, false, false);
        c11 = __builtin_amdgcn_wmma_f32_16x16x32_bf16(false, a1, false, b1,
                                                      (short)0, c11, false, false);
    }

    // C/D layout: VGPR r -> M = r + 8*half, N = l16
#pragma unroll
    for (int r = 0; r < 8; ++r) {
        int row0 = tileM + half * 8 + r;
        int row1 = row0 + 16;
        C[(size_t)row0 * ldc + tileN + l16]      = c00[r];
        C[(size_t)row0 * ldc + tileN + 16 + l16] = c01[r];
        C[(size_t)row1 * ldc + tileN + l16]      = c10[r];
        C[(size_t)row1 * ldc + tileN + 16 + l16] = c11[r];
    }
}

// =====================================================================
// Elementwise f32 -> bf16 conversion (weights / activations staging)
// =====================================================================
__global__ __launch_bounds__(256)
void cvt_bf16_kernel(const float* __restrict__ in, __bf16* __restrict__ out,
                     size_t n)
{
    size_t i = (size_t)blockIdx.x * blockDim.x + threadIdx.x;
    if (i < n) out[i] = (__bf16)in[i];
}

// Extract proj[:, 0:DTRANK] -> packed bf16 [ROWS x DTRANK]
__global__ __launch_bounds__(256)
void extract_dtraw_kernel(const float* __restrict__ proj, __bf16* __restrict__ out)
{
    int i = blockIdx.x * blockDim.x + threadIdx.x;
    if (i >= ROWS * DTRANK) return;
    int row = i / DTRANK, col = i % DTRANK;
    out[i] = (__bf16)proj[(size_t)row * NPROJ + col];
}

// =====================================================================
// Depthwise causal conv (width 4) + bias + SiLU; writes f32 (for scan)
// and bf16 (for the following GEMM).
// =====================================================================
__global__ __launch_bounds__(256)
void conv_silu_kernel(const float* __restrict__ xz,
                      const float* __restrict__ conv_w,
                      const float* __restrict__ conv_b,
                      float* __restrict__ u,
                      __bf16* __restrict__ u_bf)
{
    int idx = blockIdx.x * blockDim.x + threadIdx.x;
    if (idx >= Bn * Ln * DINNER) return;
    int ch = idx % DINNER;
    int l  = (idx / DINNER) % Ln;
    int b  = idx / (DINNER * Ln);

    float acc = conv_b[ch];
#pragma unroll
    for (int j = 0; j < DCONV; ++j) {
        int ll = l - (DCONV - 1) + j;
        if (ll >= 0)
            acc += xz[(size_t)(b * Ln + ll) * (2 * DINNER) + ch] * conv_w[ch * DCONV + j];
    }
    float sig = 1.0f / (1.0f + __expf(-acc));
    float val = acc * sig;                      // SiLU
    size_t o = (size_t)(b * Ln + l) * DINNER + ch;
    u[o]    = val;
    u_bf[o] = (__bf16)val;
}

// =====================================================================
// Selective scan: sequential over L, one thread per (b, channel), all f32.
// Fuses softplus(dt), sigmoid(a_gate), D skip, SiLU(z) gate; emits bf16 y.
// =====================================================================
__global__ __launch_bounds__(256)
void scan_kernel(const float* __restrict__ dtlin,   // [ROWS, DINNER] (pre-bias)
                 const float* __restrict__ dt_b,    // [DINNER]
                 const float* __restrict__ u,       // [ROWS, DINNER]
                 const float* __restrict__ proj,    // [ROWS, NPROJ]
                 const float* __restrict__ xz,      // [ROWS, 2*DINNER]
                 const float* __restrict__ A_log,   // [DINNER, DSTATE]
                 const float* __restrict__ D_param, // [DINNER]
                 __bf16* __restrict__ y_bf)         // [ROWS, DINNER]
{
    int idx = blockIdx.x * blockDim.x + threadIdx.x;
    if (idx >= Bn * DINNER) return;
    int b = idx / DINNER;
    int c = idx % DINNER;

    float abase[DSTATE];
    float st[DSTATE];
#pragma unroll
    for (int s = 0; s < DSTATE; ++s) {
        abase[s] = -__expf(A_log[c * DSTATE + s]);
        st[s] = 0.0f;
    }
    const float dtb = dt_b[c];
    const float Dp  = D_param[c];

    for (int l = 0; l < Ln; ++l) {
        const size_t rowP = (size_t)(b * Ln + l) * NPROJ;
        const size_t rowD = (size_t)(b * Ln + l) * DINNER;

        float dtv = dtlin[rowD + c] + dtb;
        dtv = (dtv > 20.0f) ? dtv : log1pf(__expf(dtv));   // softplus
        const float uv = u[rowD + c];
        float g = proj[rowP + DTRANK + 2 * DSTATE + c];
        g = 1.0f / (1.0f + __expf(-g));                    // sigmoid
        const float gm = 0.5f + g;

        float acc = 0.0f;
#pragma unroll
        for (int s = 0; s < DSTATE; ++s) {
            const float bt = proj[rowP + DTRANK + s];
            const float ct = proj[rowP + DTRANK + DSTATE + s];
            const float decay = __expf(dtv * abase[s] * gm);
            st[s] = decay * st[s] + dtv * bt * uv;
            acc += st[s] * ct;
        }
        const float zv = xz[(size_t)(b * Ln + l) * (2 * DINNER) + DINNER + c];
        const float sz = zv / (1.0f + __expf(-zv));        // SiLU(z)
        y_bf[rowD + c] = (__bf16)((acc + Dp * uv) * sz);
    }
}

// =====================================================================
// Host-side launcher
// =====================================================================
extern "C" void kernel_launch(void* const* d_in, const int* in_sizes, int n_in,
                              void* d_out, int out_size, void* d_ws, size_t ws_size,
                              hipStream_t stream)
{
    const float* x          = (const float*)d_in[0];
    const float* in_proj_w  = (const float*)d_in[1];  // [2*DINNER, DMODEL]
    const float* conv_w     = (const float*)d_in[2];  // [DINNER, 1, DCONV]
    const float* conv_b     = (const float*)d_in[3];
    const float* x_proj_w   = (const float*)d_in[4];  // [NPROJ, DINNER]
    const float* dt_proj_w  = (const float*)d_in[5];  // [DINNER, DTRANK]
    const float* dt_proj_b  = (const float*)d_in[6];
    const float* A_log      = (const float*)d_in[7];  // [DINNER, DSTATE]
    const float* D_param    = (const float*)d_in[8];
    const float* out_proj_w = (const float*)d_in[9];  // [DMODEL, DINNER]
    float* out = (float*)d_out;                       // [B, L, DMODEL]

    // ---------------- workspace layout ----------------
    // f32 region first, then bf16 region (all sizes multiples of 16 elems,
    // so every bf16 buffer base stays 32-byte aligned for v16bf loads).
    float* ws    = (float*)d_ws;
    float* xz    = ws;                                   // [ROWS, 2*DINNER]
    float* u     = xz    + (size_t)ROWS * 2 * DINNER;    // [ROWS, DINNER]
    float* proj  = u     + (size_t)ROWS * DINNER;        // [ROWS, NPROJ]
    float* dtlin = proj  + (size_t)ROWS * NPROJ;         // [ROWS, DINNER]

    __bf16* bfbase   = (__bf16*)(dtlin + (size_t)ROWS * DINNER);
    __bf16* x_bf     = bfbase;                                 // [ROWS, DMODEL]
    __bf16* w1_bf    = x_bf   + (size_t)ROWS * DMODEL;         // [2*DINNER, DMODEL]
    __bf16* wx_bf    = w1_bf  + (size_t)2 * DINNER * DMODEL;   // [NPROJ, DINNER]
    __bf16* wdt_bf   = wx_bf  + (size_t)NPROJ * DINNER;        // [DINNER, DTRANK]
    __bf16* wo_bf    = wdt_bf + (size_t)DINNER * DTRANK;       // [DMODEL, DINNER]
    __bf16* u_bf     = wo_bf  + (size_t)DMODEL * DINNER;       // [ROWS, DINNER]
    __bf16* dtraw_bf = u_bf   + (size_t)ROWS * DINNER;         // [ROWS, DTRANK]
    __bf16* y_bf     = dtraw_bf + (size_t)ROWS * DTRANK;       // [ROWS, DINNER]

    auto cvt = [&](const float* src, __bf16* dst, size_t n) {
        cvt_bf16_kernel<<<(int)((n + 255) / 256), 256, 0, stream>>>(src, dst, n);
    };

    // ---------------- bf16 staging (once per launch) ----------------
    cvt(x,          x_bf,   (size_t)ROWS * DMODEL);
    cvt(in_proj_w,  w1_bf,  (size_t)2 * DINNER * DMODEL);
    cvt(x_proj_w,   wx_bf,  (size_t)NPROJ * DINNER);
    cvt(dt_proj_w,  wdt_bf, (size_t)DINNER * DTRANK);
    cvt(out_proj_w, wo_bf,  (size_t)DMODEL * DINNER);

    const int wavesPerBlock = 8;
    const int blockT = 32 * wavesPerBlock;
    auto gemmBlocks = [&](int M, int N) {
        int waves = (M >> 5) * (N >> 5);
        return (waves + wavesPerBlock - 1) / wavesPerBlock;
    };

    // 1) xz = x @ in_proj_w^T   (M=2048, N=4096, K=1024)
    wmma_gemm_bf16<<<gemmBlocks(ROWS, 2 * DINNER), blockT, 0, stream>>>(
        x_bf, DMODEL, w1_bf, DMODEL, xz, 2 * DINNER, ROWS, 2 * DINNER, DMODEL);

    // 2) u = SiLU(causal depthwise conv(xb) + bias)  (f32 + bf16 copies)
    {
        int n = Bn * Ln * DINNER;
        conv_silu_kernel<<<(n + 255) / 256, 256, 0, stream>>>(
            xz, conv_w, conv_b, u, u_bf);
    }

    // 3) proj = u @ x_proj_w^T  (M=2048, N=2144, K=2048)
    wmma_gemm_bf16<<<gemmBlocks(ROWS, NPROJ), blockT, 0, stream>>>(
        u_bf, DINNER, wx_bf, DINNER, proj, NPROJ, ROWS, NPROJ, DINNER);

    // 3b) dt_raw = proj[:, :DTRANK] packed to bf16
    {
        int n = ROWS * DTRANK;
        extract_dtraw_kernel<<<(n + 255) / 256, 256, 0, stream>>>(proj, dtraw_bf);
    }

    // 4) dtlin = dt_raw @ dt_proj_w^T  (M=2048, N=2048, K=64)
    wmma_gemm_bf16<<<gemmBlocks(ROWS, DINNER), blockT, 0, stream>>>(
        dtraw_bf, DTRANK, wdt_bf, DTRANK, dtlin, DINNER, ROWS, DINNER, DTRANK);

    // 5) selective scan (f32) -> y (bf16)
    {
        int n = Bn * DINNER;
        scan_kernel<<<(n + 255) / 256, 256, 0, stream>>>(
            dtlin, dt_proj_b, u, proj, xz, A_log, D_param, y_bf);
    }

    // 6) out = y @ out_proj_w^T (M=2048, N=1024, K=2048)
    wmma_gemm_bf16<<<gemmBlocks(ROWS, DMODEL), blockT, 0, stream>>>(
        y_bf, DINNER, wo_bf, DINNER, out, DMODEL, ROWS, DMODEL, DINNER);
}
